// NonLocalBlock_61057255080213
// MI455X (gfx1250) — matrix operand order repaired and checked by reading, not compiled, not run on hardware
//
#include <hip/hip_runtime.h>

// ---------------------------------------------------------------------------
// CDNA5 / gfx1250 implementation. wave32. All GEMM-shaped work goes through
// v_wmma_f32_16x16x32_bf16 (bf16 A/B, f32 accumulate). Elementwise gating,
// softmax exp/normalization and final small convs stay f32.
// This revision: LDS halo staging for the dominant 3x3-conv kernel and
// LDS-transposed WMMA epilogues so all big output streams use contiguous
// >=16B global stores instead of scattered b16 stores.
// ---------------------------------------------------------------------------

typedef __bf16 bf16;
typedef __attribute__((ext_vector_type(16))) __bf16 v16bf;
typedef __attribute__((ext_vector_type(8)))  __bf16 v8bf;
typedef __attribute__((ext_vector_type(8)))  float  v8f;

#define DEV __device__ __forceinline__

DEV v8f fzero8() {
  v8f v;
#pragma unroll
  for (int i = 0; i < 8; ++i) v[i] = 0.0f;
  return v;
}

// A-matrix 16x32 bf16 fragment from row-major src (ld in elements).
// lane L<16 -> row L, K={0..15}; lane L>=16 -> row L-16,
// K={16..31}. Two contiguous 16B loads per lane.
DEV v16bf load_a(const bf16* base, int ld, int lane) {
  const bf16* q = base + (lane & 15) * ld + (lane >> 4) * 16;
  v8bf lo = *(const v8bf*)(q);
  v8bf hi = *(const v8bf*)(q + 8);
  v16bf f;
#pragma unroll
  for (int i = 0; i < 8; ++i) { f[i] = lo[i]; f[i + 8] = hi[i]; }
  return f;
}

// B-matrix 32x16 bf16 fragment, sourced from B^T stored row-major
// (i.e. src[n][k] contiguous in k). Lane L -> column (L&15),
// K half = 16*(L>>4). Two contiguous 16B loads per lane.
DEV v16bf load_b(const bf16* base, int ld, int lane) {
  const bf16* q = base + (lane & 15) * ld + (lane >> 4) * 16;
  v8bf lo = *(const v8bf*)(q);
  v8bf hi = *(const v8bf*)(q + 8);
  v16bf f;
#pragma unroll
  for (int i = 0; i < 8; ++i) { f[i] = lo[i]; f[i + 8] = hi[i]; }
  return f;
}

DEV v16bf pack16(v8bf lo, v8bf hi) {
  v16bf f;
#pragma unroll
  for (int i = 0; i < 8; ++i) { f[i] = lo[i]; f[i + 8] = hi[i]; }
  return f;
}

DEV v8f wmma_bf16(v16bf a, v16bf b, v8f c) {
  return __builtin_amdgcn_wmma_f32_16x16x32_bf16(
      false, a, false, b, (short)0, c, false, false);
}

DEV float sigmoidf(float x) { return 1.0f / (1.0f + __expf(-x)); }

// ---------------------------------------------------------------------------
// K0: repack conv weights to bf16. Wcat[co][tap*128+ci], co: 0..127=Wadd,
// 128..191=Wdot, 192..255=Wout. Also Wfus -> bf16.
// ---------------------------------------------------------------------------
__global__ void k_prep_w(const float* __restrict__ Wadd, const float* __restrict__ Wdot,
                         const float* __restrict__ Wout, const float* __restrict__ Wfus,
                         bf16* __restrict__ Wcat, bf16* __restrict__ Wfusb) {
  int tid = blockIdx.x * blockDim.x + threadIdx.x;
  const int NC = 256 * 1152;
  if (tid < NC) {
    int co = tid / 1152, k = tid % 1152;
    int tap = k / 128, ci = k % 128;
    float v;
    if (co < 128)      v = Wadd[((co)       * 128 + ci) * 9 + tap];
    else if (co < 192) v = Wdot[((co - 128) * 128 + ci) * 9 + tap];
    else               v = Wout[((co - 192) * 128 + ci) * 9 + tap];
    Wcat[co * 1152 + k] = (bf16)v;
  } else if (tid < NC + 64 * 128) {
    int i = tid - NC;
    Wfusb[i] = (bf16)Wfus[i];
  }
}

// ---------------------------------------------------------------------------
// K1: theta/phi/g 1x1 convs (tiny: 0.1 GFLOP). Channel-major bf16 outputs;
// the raw torch reshape makes these flat buffers directly the [1024,32]
// row-major GEMM operands (stride 32).
// ---------------------------------------------------------------------------
__global__ void k_qkv(const float* __restrict__ x,
                      const float* __restrict__ Wt, const float* __restrict__ bt,
                      const float* __restrict__ Wp, const float* __restrict__ bp,
                      const float* __restrict__ Wg, const float* __restrict__ bg,
                      bf16* __restrict__ tb, bf16* __restrict__ pb, bf16* __restrict__ gb) {
  int tid = blockIdx.x * blockDim.x + threadIdx.x;
  if (tid >= 8 * 3 * 32 * 1024) return;
  int sp = tid & 1023;
  int ch = (tid >> 10) & 31;
  int which = (tid >> 15) % 3;
  int b = tid / 98304;
  const float* W; const float* bias; bf16* out;
  if (which == 0)      { W = Wt; bias = bt; out = tb; }
  else if (which == 1) { W = Wp; bias = bp; out = pb; }
  else                 { W = Wg; bias = bg; out = gb; }
  const float* xp = x + b * 65536 + sp;
  const float* wp = W + ch * 64;
  float acc = bias[ch];
#pragma unroll
  for (int k = 0; k < 64; ++k) acc += wp[k] * xp[k * 1024];
  out[b * 32768 + ch * 1024 + sp] = (bf16)acc;
}

// K2: gT[b][c][m] = g_x[m][c]  (so GEMM3 B-fragments are contiguous)
__global__ void k_transpose_g(const bf16* __restrict__ gb, bf16* __restrict__ gT) {
  int tid = blockIdx.x * blockDim.x + threadIdx.x;
  if (tid >= 262144) return;
  int m = tid & 1023;
  int c = (tid >> 10) & 31;
  int b = tid >> 15;
  gT[b * 32768 + c * 1024 + m] = gb[b * 32768 + m * 32 + c];
}

// ---------------------------------------------------------------------------
// K3: f = theta_x @ phi_x^T  [8 x 1024 x 1024], K=32 -> one WMMA per tile.
// One wave computes a 16x64 strip (A reused over 4 B tiles). D tiles are
// transposed through per-wave LDS so the 16MB f stream is written with
// contiguous b128 stores.
// ---------------------------------------------------------------------------
__global__ void __launch_bounds__(256) k_gemm_f(const bf16* __restrict__ tb,
                                                const bf16* __restrict__ pb,
                                                bf16* __restrict__ fbuf) {
  __shared__ bf16 xp[8][16 * 64];    // per-wave [16 n][64 m] tile
  int lane = threadIdx.x & 31;
  int wv = threadIdx.x >> 5;
  int gid = blockIdx.x * 8 + wv;     // 8192 waves
  int b = gid >> 10;
  int rem = gid & 1023;
  int nt = rem >> 4;                 // 0..63 n-tile
  int mg = rem & 15;                 // 0..15, 4 m-tiles each
  int nl = lane & 15, hi = lane >> 4;
  v16bf a = load_a(tb + b * 32768 + nt * 16 * 32, 32, lane);
  bf16* x0 = xp[wv];
#pragma unroll
  for (int j = 0; j < 4; ++j) {
    int m0 = mg * 64 + j * 16;
    v16bf bf = load_b(pb + b * 32768 + m0 * 32, 32, lane);
    v8f acc = fzero8();
    acc = wmma_bf16(a, bf, acc);
#pragma unroll
    for (int r = 0; r < 8; ++r)
      x0[(r + hi * 8) * 64 + j * 16 + nl] = (bf16)acc[r];
  }
  __syncthreads();
  bf16* out = fbuf + (size_t)b * 1048576 + nt * 16 * 1024 + mg * 64;
#pragma unroll
  for (int i = 0; i < 4; ++i) {
    int c = lane + 32 * i;           // 128 16B-chunks: row = c/8, part = c%8
    int row = c >> 3, part = c & 7;
    *(v8bf*)(out + row * 1024 + part * 8) = *(const v8bf*)(x0 + row * 64 + part * 8);
  }
}

// ---------------------------------------------------------------------------
// K4: stacked NHWC bf16 [8][128y][128x][128ch]: ch 0..63 = lstm_in (= f flat,
// raw reshape), ch 64..127 = last_h. NHWC makes conv B-fragments contiguous.
// ---------------------------------------------------------------------------
__global__ void k_stacked(const bf16* __restrict__ fbuf, const float* __restrict__ h,
                          bf16* __restrict__ stk) {
  int tid = blockIdx.x * blockDim.x + threadIdx.x;
  if (tid >= 16777216) return;
  int cc = tid & 127;
  int ps = tid >> 7;                 // b*16384 + sp
  int b = ps >> 14;
  int sp = ps & 16383;
  bf16 v;
  if (cc < 64) v = fbuf[(size_t)b * 1048576 + cc * 16384 + sp];
  else         v = (bf16)h[(size_t)b * 1048576 + (cc - 64) * 16384 + sp];
  stk[tid] = v;
}

// ---------------------------------------------------------------------------
// K5: fused 3x3 convs (add/dot/out gates) as implicit GEMM.
// M=256 co, N=spatial, K=1152 (tap-major). ~77 GFLOP total.
// The 3x18x128ci bf16 halo tile (13.8KB) is staged once per block into LDS
// (coalesced 256B segments, zero-fill padding), shared by all 4 waves; the
// 36-step WMMA loop reads B fragments as aligned ds_load_b128 pairs.
// Outputs are transposed through LDS for 32B-contiguous global stores.
// ---------------------------------------------------------------------------
__global__ void __launch_bounds__(128) k_conv3(const bf16* __restrict__ Wcat,
                        const bf16* __restrict__ stk,
                        const float* __restrict__ badd, const float* __restrict__ bdot,
                        const float* __restrict__ bout, bf16* __restrict__ gates) {
  __shared__ unsigned int tileu[3456];   // halo [3 rows][18 x][128 ci] bf16
  __shared__ bf16 xp[4][64 * 16];        // per-wave [64 co][16 sp] transpose
  int tid = threadIdx.x;
  int lane = tid & 31;
  int wv = tid >> 5;                     // 0..3, co group = wv*64
  int xt = blockIdx.x;                   // 0..7
  int y  = blockIdx.y;                   // 0..127
  int b  = blockIdx.z;                   // 0..7

  // ---- stage halo (zero-padded) ----
  const unsigned int* sg = (const unsigned int*)stk;  // 2 bf16 per dword
#pragma unroll 1
  for (int idx = tid; idx < 3456; idx += 128) {
    int seg = idx >> 6;                  // 0..53 -> (row, xcol)
    int d   = idx & 63;                  // dword within the 128-ci vector
    int r = seg / 18, xc = seg % 18;
    int yy = y - 1 + r;
    int xx = xt * 16 - 1 + xc;
    unsigned int v = 0u;
    if (((unsigned)yy < 128u) && ((unsigned)xx < 128u))
      v = sg[(size_t)(b * 16384 + yy * 128 + xx) * 64 + d];
    tileu[idx] = v;
  }
  __syncthreads();
  const bf16* tile = (const bf16*)tileu;

  int nl = lane & 15, hi = lane >> 4;
  int co_base = wv * 64;
  v8f acc[4];
#pragma unroll
  for (int j = 0; j < 4; ++j) acc[j] = fzero8();

#pragma unroll 1
  for (int tap = 0; tap < 9; ++tap) {
    int dy = tap / 3, dx = tap % 3;      // 0..2, indexes halo directly
    const bf16* brow = tile + (dy * 18 + nl + dx) * 128 + hi * 16;
#pragma unroll
    for (int cb = 0; cb < 4; ++cb) {
      v8bf lo = *(const v8bf*)(brow + cb * 32);
      v8bf up = *(const v8bf*)(brow + cb * 32 + 8);
      v16bf bf = pack16(lo, up);
      int k0 = tap * 128 + cb * 32;
#pragma unroll
      for (int j = 0; j < 4; ++j) {
        v16bf a = load_a(Wcat + (co_base + j * 16) * 1152 + k0, 1152, lane);
        acc[j] = wmma_bf16(a, bf, acc[j]);
      }
    }
  }

  // ---- bias + transpose through LDS, then 32B-contiguous stores ----
  bf16* x0 = xp[wv];
#pragma unroll
  for (int j = 0; j < 4; ++j) {
#pragma unroll
    for (int r = 0; r < 8; ++r) {
      int cl = j * 16 + r + hi * 8;      // local co 0..63
      int co = co_base + cl;
      float bias = (co < 128) ? badd[co] : ((co < 192) ? bdot[co - 128] : bout[co - 192]);
      x0[cl * 16 + nl] = (bf16)(acc[j][r] + bias);
    }
  }
  __syncthreads();
  int sp0 = y * 128 + xt * 16;
#pragma unroll
  for (int i = 0; i < 2; ++i) {
    int cl = lane * 2 + i;               // two co rows per lane, 32B each
    int co = co_base + cl;
    const v8bf* s = (const v8bf*)(x0 + cl * 16);
    bf16* dst = gates + ((size_t)b * 256 + co) * 16384 + sp0;
    *(v8bf*)(dst)     = s[0];
    *(v8bf*)(dst + 8) = s[1];
  }
}

// ---------------------------------------------------------------------------
// K6: LSTM gating (f32). Writes next_h/next_c to d_out, and hidden (bf16)
// into stacked channels 64..127 for the fusion 1x1 conv.
// ---------------------------------------------------------------------------
__global__ void k_gates(const bf16* __restrict__ gates, const float* __restrict__ last_c,
                        float* __restrict__ next_h, float* __restrict__ next_c,
                        bf16* __restrict__ stk) {
  int tid = blockIdx.x * blockDim.x + threadIdx.x;
  if (tid >= 8388608) return;
  int sp = tid & 16383;
  int ch = (tid >> 14) & 63;
  int b = tid >> 20;
  size_t base = (size_t)b * 256 * 16384;
  float ing = (float)gates[base + (ch      ) * 16384 + sp];
  float cg  = (float)gates[base + (ch +  64) * 16384 + sp];
  float rem = (float)gates[base + (ch + 128) * 16384 + sp];
  float og  = (float)gates[base + (ch + 192) * 16384 + sp];
  float pc = last_c[(size_t)b * 1048576 + ch * 16384 + sp];
  float cell = pc + sigmoidf(ing) * tanhf(cg);
  float o_a = og > 0.0f ? og : 0.1f * og;         // leaky_relu 0.1
  float hid = o_a * tanhf(cell);
  size_t oi = (size_t)b * 1048576 + ch * 16384 + sp;
  next_h[oi] = hid;
  next_c[oi] = sigmoidf(rem) * cell;
  stk[((size_t)b * 16384 + sp) * 128 + 64 + ch] = (bf16)hid;
}

// ---------------------------------------------------------------------------
// K7: fusion 1x1 conv = GEMM [64 x 128] x [128 x 16384] per batch, then the
// max-free softmax numerator exp() written bf16 (reuses f buffer).
// LDS-transposed epilogue -> 32B-contiguous stores.
// ---------------------------------------------------------------------------
__global__ void __launch_bounds__(256) k_gemm_fused(const bf16* __restrict__ Wfusb,
                             const float* __restrict__ bfus,
                             const bf16* __restrict__ stk, bf16* __restrict__ ef) {
  __shared__ bf16 xp[8][64 * 16];
  int lane = threadIdx.x & 31;
  int wv = threadIdx.x >> 5;
  int gid = blockIdx.x * 8 + wv;     // 8192 waves
  int b = gid >> 10;
  int st = gid & 1023;
  int sp0 = st * 16;
  int nl = lane & 15, hi = lane >> 4;
  v8f acc[4];
#pragma unroll
  for (int j = 0; j < 4; ++j) acc[j] = fzero8();
  const bf16* Bbase = stk + ((size_t)b * 16384 + sp0) * 128;
#pragma unroll
  for (int k0 = 0; k0 < 128; k0 += 32) {
    v16bf bf = load_b(Bbase + k0, 128, lane);
#pragma unroll
    for (int j = 0; j < 4; ++j)
      acc[j] = wmma_bf16(load_a(Wfusb + j * 16 * 128 + k0, 128, lane), bf, acc[j]);
  }
  bf16* x0 = xp[wv];
#pragma unroll
  for (int j = 0; j < 4; ++j) {
#pragma unroll
    for (int r = 0; r < 8; ++r) {
      int cl = j * 16 + r + hi * 8;  // co 0..63
      x0[cl * 16 + nl] = (bf16)__expf(acc[j][r] + bfus[cl]);
    }
  }
  __syncthreads();
  bf16* out = ef + (size_t)b * 1048576 + sp0;   // f2 flat == co*16384 + sp
#pragma unroll
  for (int i = 0; i < 2; ++i) {
    int cl = lane * 2 + i;
    const v8bf* s = (const v8bf*)(x0 + cl * 16);
    bf16* dst = out + (size_t)cl * 16384;
    *(v8bf*)(dst)     = s[0];
    *(v8bf*)(dst + 8) = s[1];
  }
}

// K8: softmax denominators: rowsum over 1024 ef values per (b,n). Wave/row.
__global__ void k_rowsum(const bf16* __restrict__ ef, float* __restrict__ rs) {
  int lane = threadIdx.x & 31;
  int wv = threadIdx.x >> 5;
  int row = blockIdx.x * 8 + wv;     // 8192 rows
  const bf16* p = ef + (size_t)row * 1024;
  float s = 0.0f;
  for (int i = lane; i < 1024; i += 32) s += (float)p[i];
#pragma unroll
  for (int m = 16; m > 0; m >>= 1) s += __shfl_xor(s, m, 32);
  if (lane == 0) rs[row] = s;
}

// ---------------------------------------------------------------------------
// K9: y = softmax(f2) @ g_x : per batch [1024 x 1024] x [1024 x 32].
// A = ef (row-major), B columns from gT (contiguous), scale by 1/rowsum.
// Output stored flat [n*32+c] (f32) == raw reshape to [32][1024] for K10.
// ---------------------------------------------------------------------------
__global__ void k_gemm_y(const bf16* __restrict__ ef, const bf16* __restrict__ gT,
                         const float* __restrict__ rs, float* __restrict__ ybuf) {
  int lane = threadIdx.x & 31;
  int wv = threadIdx.x >> 5;
  int gid = blockIdx.x * 8 + wv;     // 512 waves
  int b = gid >> 6;
  int nt = gid & 63;
  int nl = lane & 15, hi = lane >> 4;
  v8f acc[2];
  acc[0] = fzero8(); acc[1] = fzero8();
  const bf16* A  = ef + (size_t)b * 1048576 + nt * 16 * 1024;
  const bf16* B0 = gT + (size_t)b * 32768;
#pragma unroll 4
  for (int k0 = 0; k0 < 1024; k0 += 32) {
    v16bf a = load_a(A + k0, 1024, lane);
    acc[0] = wmma_bf16(a, load_b(B0 + k0, 1024, lane), acc[0]);
    acc[1] = wmma_bf16(a, load_b(B0 + 16 * 1024 + k0, 1024, lane), acc[1]);
  }
  float* out = ybuf + (size_t)b * 32768;
#pragma unroll
  for (int r = 0; r < 8; ++r) {
    int n = nt * 16 + r + hi * 8;
    float inv = 1.0f / rs[b * 1024 + n];
    out[n * 32 +      nl] = acc[0][r] * inv;
    out[n * 32 + 16 + nl] = acc[1][r] * inv;
  }
}

// K10: final 1x1 conv z = Wy @ y (f32 VALU, 0.03 GFLOP) -> d_out.
__global__ void k_zconv(const float* __restrict__ ybuf, const float* __restrict__ Wy,
                        const float* __restrict__ by, float* __restrict__ z) {
  int tid = blockIdx.x * blockDim.x + threadIdx.x;
  if (tid >= 524288) return;
  int sp = tid & 1023;
  int cz = (tid >> 10) & 63;
  int b = tid >> 16;
  const float* yp = ybuf + b * 32768 + sp;   // y viewed [32][1024] via raw reshape
  const float* wp = Wy + cz * 32;
  float acc = by[cz];
#pragma unroll
  for (int c = 0; c < 32; ++c) acc += wp[c] * yp[c * 1024];
  z[b * 65536 + cz * 1024 + sp] = acc;
}

// ---------------------------------------------------------------------------
extern "C" void kernel_launch(void* const* d_in, const int* in_sizes, int n_in,
                              void* d_out, int out_size, void* d_ws, size_t ws_size,
                              hipStream_t stream) {
  const float* x      = (const float*)d_in[0];
  const float* last_h = (const float*)d_in[1];
  const float* last_c = (const float*)d_in[2];
  const float* Wg     = (const float*)d_in[3];
  const float* bg     = (const float*)d_in[4];
  const float* Wphi   = (const float*)d_in[5];
  const float* bphi   = (const float*)d_in[6];
  const float* Wtheta = (const float*)d_in[7];
  const float* btheta = (const float*)d_in[8];
  const float* Wy     = (const float*)d_in[9];
  const float* by     = (const float*)d_in[10];
  const float* Wadd   = (const float*)d_in[11];
  const float* badd   = (const float*)d_in[12];
  const float* Wdot   = (const float*)d_in[13];
  const float* bdot   = (const float*)d_in[14];
  const float* Wout   = (const float*)d_in[15];
  const float* bout   = (const float*)d_in[16];
  const float* Wfus   = (const float*)d_in[17];
  const float* bfus   = (const float*)d_in[18];

  float* z      = (float*)d_out;               // [8,64,32,32]
  float* next_h = z + 524288;                  // [8,64,128,128]
  float* next_c = next_h + 8388608;            // [8,64,128,128]

  char* w = (char*)d_ws;
  size_t off = 0;
  auto take = [&](size_t nbytes) -> void* {
    void* p = w + off;
    off += (nbytes + 255) & ~(size_t)255;
    return p;
  };
  bf16*  tb    = (bf16*)take((size_t)8 * 32768 * 2);         // theta (ch-major)
  bf16*  pbuf  = (bf16*)take((size_t)8 * 32768 * 2);         // phi
  bf16*  gb    = (bf16*)take((size_t)8 * 32768 * 2);         // g
  bf16*  gT    = (bf16*)take((size_t)8 * 32768 * 2);         // g_x transposed
  bf16*  Wcat  = (bf16*)take((size_t)256 * 1152 * 2);        // repacked 3x3 weights
  bf16*  Wfusb = (bf16*)take((size_t)64 * 128 * 2);
  bf16*  fbuf  = (bf16*)take((size_t)8 * 1048576 * 2);       // f, later reused as ef
  bf16*  stk   = (bf16*)take((size_t)8 * 16384 * 128 * 2);   // NHWC stacked / cat
  bf16*  gates = (bf16*)take((size_t)8 * 256 * 16384 * 2);   // raw gate pre-acts
  float* rs    = (float*)take((size_t)8192 * 4);             // softmax row sums
  float* ybuf  = (float*)take((size_t)8 * 32768 * 4);        // y (f32)
  bf16*  ef    = fbuf;   // f dead after k_stacked; reuse for exp(fused)

  k_prep_w<<<(256 * 1152 + 64 * 128 + 255) / 256, 256, 0, stream>>>(
      Wadd, Wdot, Wout, Wfus, Wcat, Wfusb);
  k_qkv<<<786432 / 256, 256, 0, stream>>>(
      x, Wtheta, btheta, Wphi, bphi, Wg, bg, tb, pbuf, gb);
  k_transpose_g<<<262144 / 256, 256, 0, stream>>>(gb, gT);
  k_gemm_f<<<1024, 256, 0, stream>>>(tb, pbuf, fbuf);
  k_stacked<<<16777216 / 256, 256, 0, stream>>>(fbuf, last_h, stk);
  k_conv3<<<dim3(8, 128, 8), 128, 0, stream>>>(Wcat, stk, badd, bdot, bout, gates);
  k_gates<<<8388608 / 256, 256, 0, stream>>>(gates, last_c, next_h, next_c, stk);
  k_gemm_fused<<<1024, 256, 0, stream>>>(Wfusb, bfus, stk, ef);
  k_rowsum<<<1024, 256, 0, stream>>>(ef, rs);
  k_gemm_y<<<64, 256, 0, stream>>>(ef, gT, rs, ybuf);
  k_zconv<<<524288 / 256, 256, 0, stream>>>(ybuf, Wy, by, z);
}
